// BidirectionalGRU_41566693491037
// MI455X (gfx1250) — matrix operand order repaired and checked by reading, not compiled, not run on hardware
//
#include <hip/hip_runtime.h>

// ---------------------------------------------------------------------------
// MI455X (gfx1250) bidirectional GRU.
//  - bf16 WMMA (v_wmma_f32_16x16x32_bf16) everywhere, f32 accumulate.
//  - Software-pipelined fragment loads (double-buffered regs) so WMMAs overlap
//    L2 latency instead of s_wait_loadcnt 0 before every matrix op.
//  - Recurrence: persistent 512-thread workgroup per direction; h double-
//    buffered in dynamic LDS -> ONE barrier per timestep; W_hh streamed from
//    the 192 MB L2 every step (1.5 MB bf16).
//  - Layer-1 backward runs exactly 1 step (only out[:, -1, :] is consumed).
// ---------------------------------------------------------------------------

typedef __bf16 v16bf __attribute__((ext_vector_type(16)));
typedef float  v8f   __attribute__((ext_vector_type(8)));
typedef float  fvec4 __attribute__((ext_vector_type(4)));

union frag16 { fvec4 f[2]; v16bf v; };

// A frag: base already includes +16*(lane/16); 16 contiguous halfs (32 B).
__device__ __forceinline__ void load_a_frag(frag16& fr, const __bf16* p) {
  fr.f[0] = *(const fvec4*)(p);
  fr.f[1] = *(const fvec4*)(p + 8);
}
// B frag: base already includes +16*(lane/16); 16 contiguous halfs (32 B).
__device__ __forceinline__ void load_b_frag(frag16& fr, const __bf16* p) {
  fr.f[0] = *(const fvec4*)(p);
  fr.f[1] = *(const fvec4*)(p + 8);
}
__device__ __forceinline__ v8f wmma_bf16(const frag16& a, const frag16& b, v8f c) {
  return __builtin_amdgcn_wmma_f32_16x16x32_bf16(false, a.v, false, b.v,
                                                 (short)0, c, false, false);
}

// ---------------------------------------------------------------------------
// C[M,N] = A[M,K] (row-major bf16) x W[N,K]^T (row-major bf16) + bias
// One wave: 16x64 tile (4 accumulators, shared A frag), 2-stage pipelined.
// ---------------------------------------------------------------------------
__global__ __launch_bounds__(256)
void wmma_gemm_bf16(const __bf16* __restrict__ A, int lda,
                    const __bf16* __restrict__ Bw, int ldb,
                    const float* __restrict__ bias,
                    float* __restrict__ C, int ldc,
                    int M, int N, int K) {
  const int lane = threadIdx.x & 31;
  const int wid  = blockIdx.x * (blockDim.x >> 5) + (threadIdx.x >> 5);
  const int nt64 = N >> 6;
  const int mt   = wid / nt64;
  const int nt   = wid % nt64;
  if (mt * 16 >= M) return;
  const int row0 = mt << 4;
  const int n0   = nt << 6;
  const int lm = lane & 15, lh = lane >> 4;

  v8f acc[4] = {};
  const __bf16* pa = A  + (size_t)(row0 + lm) * lda + 16 * lh;
  const __bf16* pb = Bw + (size_t)(n0 + lm) * ldb + 16 * lh;
  const size_t bstep = (size_t)16 * ldb;

  frag16 ua0, ua1, ub0[4], ub1[4];
  load_a_frag(ua0, pa);
#pragma unroll
  for (int j = 0; j < 4; ++j) load_b_frag(ub0[j], pb + j * bstep);

  for (int k = 0; k < K; k += 64) {
    if (k + 32 < K) {
      load_a_frag(ua1, pa + k + 32);
#pragma unroll
      for (int j = 0; j < 4; ++j) load_b_frag(ub1[j], pb + j * bstep + k + 32);
    }
#pragma unroll
    for (int j = 0; j < 4; ++j) acc[j] = wmma_bf16(ua0, ub0[j], acc[j]);

    if (k + 64 < K) {
      load_a_frag(ua0, pa + k + 64);
#pragma unroll
      for (int j = 0; j < 4; ++j) load_b_frag(ub0[j], pb + j * bstep + k + 64);
    }
#pragma unroll
    for (int j = 0; j < 4; ++j) acc[j] = wmma_bf16(ua1, ub1[j], acc[j]);
  }

#pragma unroll
  for (int j = 0; j < 4; ++j) {
    const int n = n0 + j * 16 + lm;
    const float bv = bias ? bias[n] : 0.0f;
#pragma unroll
    for (int g = 0; g < 8; ++g) {
      const int m = row0 + 8 * lh + g;   // C/D layout: row = 8*(lane/16)+g
      C[(size_t)m * ldc + n] = acc[j][g] + bv;
    }
  }
}

// ---------------------------------------------------------------------------
// Persistent GRU recurrence. blockIdx.x = direction (0 fwd, 1 bwd).
// 512 threads = 16 waves; wave w owns h-columns [32w, 32w+32).
// h is double-buffered in dynamic LDS -> single barrier per timestep.
// gx row index = b*gxBS + t*gxTS (lets layer-1 bwd use a 32-row gx).
// ---------------------------------------------------------------------------
#define H_STRIDE 520   // +8 halfs -> 4-bank rotation/row, conflict-free b128
#define H_BUFSZ  (32 * H_STRIDE)

__global__ __launch_bounds__(512)
void gru_rec(const float* __restrict__ gx_f, const float* __restrict__ gx_b,
             const __bf16* __restrict__ whh_f, const __bf16* __restrict__ whh_b,
             const float* __restrict__ bhh_f, const float* __restrict__ bhh_b,
             int T, int nsteps_f, int nsteps_b,
             int gxBS_f, int gxTS_f, int gxBS_b, int gxTS_b,
             __bf16* __restrict__ outSeq,    // [B*T, 1024] bf16 or null
             float*  __restrict__ outFinal)  // [B, 1024] f32 or null
{
  extern __shared__ __bf16 hbuf[];           // 2 x [32][H_STRIDE]

  const int dir = blockIdx.x;
  const float*  gx   = dir ? gx_b  : gx_f;
  const __bf16* whh  = dir ? whh_b : whh_f;
  const float*  bhh  = dir ? bhh_b : bhh_f;
  const int nsteps   = dir ? nsteps_b : nsteps_f;
  const int gxBS     = dir ? gxBS_b : gxBS_f;
  const int gxTS     = dir ? gxTS_b : gxTS_f;
  const int doff     = dir * 512;

  const int tid = threadIdx.x;
  const int lane = tid & 31, wv = tid >> 5;
  const int lm = lane & 15, lh = lane >> 4;

  for (int i = tid; i < H_BUFSZ; i += 512) hbuf[i] = (__bf16)0.0f;  // buffer 0
  __syncthreads();

  for (int s = 0; s < nsteps; ++s) {
    const int t = dir ? (T - 1 - s) : s;
    const __bf16* hc = hbuf + (s & 1) * H_BUFSZ;        // read current h
    __bf16*       hn = hbuf + ((s & 1) ^ 1) * H_BUFSZ;  // write next h

#pragma unroll
    for (int sub = 0; sub < 2; ++sub) {
      const int j = wv * 32 + sub * 16 + lm;     // h column this lane owns
      const __bf16* wR = whh + (size_t)(       j) * 512 + 16 * lh;
      const __bf16* wZ = whh + (size_t)( 512 + j) * 512 + 16 * lh;
      const __bf16* wN = whh + (size_t)(1024 + j) * 512 + 16 * lh;
      const float bR = bhh[j], bZ = bhh[512 + j], bN = bhh[1024 + j];

#pragma unroll
      for (int mt = 0; mt < 2; ++mt) {
        v8f aR = {}, aZ = {}, aN = {};
        const __bf16* pa = hc + (mt * 16 + lm) * H_STRIDE + 16 * lh;

        frag16 ua, bR0, bZ0, bN0, bR1, bZ1, bN1;
        load_b_frag(bR0, wR); load_b_frag(bZ0, wZ); load_b_frag(bN0, wN);

        for (int k = 0; k < 512; k += 64) {
          // stage 1: prefetch k+32 weights, compute on slot 0
          load_b_frag(bR1, wR + k + 32);
          load_b_frag(bZ1, wZ + k + 32);
          load_b_frag(bN1, wN + k + 32);
          load_a_frag(ua, pa + k);               // ds_load_b128 (fast)
          aR = wmma_bf16(ua, bR0, aR);
          aZ = wmma_bf16(ua, bZ0, aZ);
          aN = wmma_bf16(ua, bN0, aN);
          // stage 2: prefetch k+64 weights, compute on slot 1
          if (k + 64 < 512) {
            load_b_frag(bR0, wR + k + 64);
            load_b_frag(bZ0, wZ + k + 64);
            load_b_frag(bN0, wN + k + 64);
          }
          load_a_frag(ua, pa + k + 32);
          aR = wmma_bf16(ua, bR1, aR);
          aZ = wmma_bf16(ua, bZ1, aZ);
          aN = wmma_bf16(ua, bN1, aN);
        }

#pragma unroll
        for (int g = 0; g < 8; ++g) {
          const int b = mt * 16 + 8 * lh + g;    // batch row of this element
          const float hold_v = (float)hc[b * H_STRIDE + j];
          const size_t grow = ((size_t)b * gxBS + (size_t)t * gxTS) * 1536;
          const float xr = gx[grow + j];
          const float xz = gx[grow + 512 + j];
          const float xn = gx[grow + 1024 + j];
          const float rr = 1.0f / (1.0f + __expf(-(xr + aR[g] + bR)));
          const float zz = 1.0f / (1.0f + __expf(-(xz + aZ[g] + bZ)));
          const float nn = tanhf(xn + rr * (aN[g] + bN));
          const float hv = (1.0f - zz) * nn + zz * hold_v;
          hn[b * H_STRIDE + j] = (__bf16)hv;     // write NEXT buffer: no race
          if (outSeq)
            outSeq[((size_t)b * T + t) * 1024 + doff + j] = (__bf16)hv;
          if (outFinal && s == nsteps - 1)
            outFinal[(size_t)b * 1024 + doff + j] = hv;
        }
      }
    }
    __syncthreads();   // next buffer fully written before anyone reads it
  }
}

__global__ void cvt_f32_bf16(const float* __restrict__ in, __bf16* __restrict__ out, int n) {
  int i = blockIdx.x * blockDim.x + threadIdx.x;
  if (i < n) out[i] = (__bf16)in[i];
}

__global__ void fc_kernel(const float* __restrict__ pooled,
                          const float* __restrict__ fw,
                          const float* __restrict__ fb,
                          float* __restrict__ out) {
  int idx = blockIdx.x * blockDim.x + threadIdx.x;
  if (idx >= 32 * 10) return;
  int b = idx / 10, c = idx % 10;
  const float* pr = pooled + (size_t)b * 1024;
  const float* wr = fw + (size_t)c * 1024;
  float s = fb[c];
  for (int k = 0; k < 1024; ++k) s += pr[k] * wr[k];
  out[idx] = s;
}

extern "C" void kernel_launch(void* const* d_in, const int* in_sizes, int n_in,
                              void* d_out, int out_size, void* d_ws, size_t ws_size,
                              hipStream_t stream) {
  (void)in_sizes; (void)n_in; (void)out_size; (void)ws_size;
  const float* x     = (const float*)d_in[0];
  const float* wih0f = (const float*)d_in[1];
  const float* whh0f = (const float*)d_in[2];
  const float* bih0f = (const float*)d_in[3];
  const float* bhh0f = (const float*)d_in[4];
  const float* wih0b = (const float*)d_in[5];
  const float* whh0b = (const float*)d_in[6];
  const float* bih0b = (const float*)d_in[7];
  const float* bhh0b = (const float*)d_in[8];
  const float* wih1f = (const float*)d_in[9];
  const float* whh1f = (const float*)d_in[10];
  const float* bih1f = (const float*)d_in[11];
  const float* bhh1f = (const float*)d_in[12];
  const float* wih1b = (const float*)d_in[13];
  const float* whh1b = (const float*)d_in[14];
  const float* bih1b = (const float*)d_in[15];
  const float* bhh1b = (const float*)d_in[16];
  const float* fcw   = (const float*)d_in[17];
  const float* fcb   = (const float*)d_in[18];
  float* out = (float*)d_out;

  constexpr int B = 32, T = 512, I = 256, H = 512;
  constexpr int BT = B * T, G3 = 3 * H;

  char* p = (char*)d_ws;
  auto carve = [&](size_t bytes) {
    char* r = p;
    p += (bytes + 255) & ~(size_t)255;
    return r;
  };
  __bf16* x_bf     = (__bf16*)carve((size_t)BT * I * 2);
  __bf16* wih0f_bf = (__bf16*)carve((size_t)G3 * I * 2);
  __bf16* wih0b_bf = (__bf16*)carve((size_t)G3 * I * 2);
  __bf16* whh0f_bf = (__bf16*)carve((size_t)G3 * H * 2);
  __bf16* whh0b_bf = (__bf16*)carve((size_t)G3 * H * 2);
  __bf16* whh1f_bf = (__bf16*)carve((size_t)G3 * H * 2);
  __bf16* whh1b_bf = (__bf16*)carve((size_t)G3 * H * 2);
  __bf16* wih1f_bf = (__bf16*)carve((size_t)G3 * 2 * H * 2);
  __bf16* wih1b_bf = (__bf16*)carve((size_t)G3 * 2 * H * 2);
  float* gxA    = (float*)carve((size_t)BT * G3 * 4);  // layer0 fwd, reused layer1 fwd
  float* gxB    = (float*)carve((size_t)BT * G3 * 4);  // layer0 bwd
  float* gx1b   = (float*)carve((size_t)B * G3 * 4);   // layer1 bwd (t=T-1 only)
  __bf16* out0  = (__bf16*)carve((size_t)BT * 2 * H * 2);
  float* pooled = (float*)carve((size_t)B * 2 * H * 4);

  auto cvt = [&](const float* src, __bf16* dst, int n) {
    cvt_f32_bf16<<<(n + 255) / 256, 256, 0, stream>>>(src, dst, n);
  };
  cvt(x, x_bf, BT * I);
  cvt(wih0f, wih0f_bf, G3 * I);
  cvt(wih0b, wih0b_bf, G3 * I);
  cvt(whh0f, whh0f_bf, G3 * H);
  cvt(whh0b, whh0b_bf, G3 * H);
  cvt(whh1f, whh1f_bf, G3 * H);
  cvt(whh1b, whh1b_bf, G3 * H);
  cvt(wih1f, wih1f_bf, G3 * 2 * H);
  cvt(wih1b, wih1b_bf, G3 * 2 * H);

  auto gemm = [&](const __bf16* A, int lda, const __bf16* Bw, int ldb,
                  const float* bias, float* C, int ldc, int M, int N, int K) {
    int tiles = (M / 16) * (N / 64);
    int blocks = (tiles + 7) / 8;
    wmma_gemm_bf16<<<blocks, 256, 0, stream>>>(A, lda, Bw, ldb, bias, C, ldc, M, N, K);
  };

  const size_t rec_lds = (size_t)2 * H_BUFSZ * sizeof(__bf16);  // 66,560 B

  // Layer 0 input projections
  gemm(x_bf, I, wih0f_bf, I, bih0f, gxA, G3, BT, G3, I);
  gemm(x_bf, I, wih0b_bf, I, bih0b, gxB, G3, BT, G3, I);

  // Layer 0 recurrence (both directions, full sequence -> out0 bf16)
  gru_rec<<<2, 512, rec_lds, stream>>>(gxA, gxB, whh0f_bf, whh0b_bf, bhh0f, bhh0b,
                                       T, T, T, T, 1, T, 1, out0, nullptr);

  // Layer 1 input projections (fwd: full; bwd: only rows at t = T-1 via strided A)
  gemm(out0, 2 * H, wih1f_bf, 2 * H, bih1f, gxA, G3, BT, G3, 2 * H);
  gemm(out0 + (size_t)(T - 1) * 2 * H, T * 2 * H, wih1b_bf, 2 * H, bih1b,
       gx1b, G3, B, G3, 2 * H);

  // Layer 1 recurrence: fwd full scan (write final h only), bwd single step
  gru_rec<<<2, 512, rec_lds, stream>>>(gxA, gx1b, whh1f_bf, whh1b_bf, bhh1f, bhh1b,
                                       T, T, 1, T, 1, 1, 0, nullptr, pooled);

  // FC head
  fc_kernel<<<2, 256, 0, stream>>>(pooled, fcw, fcb, out);
}
